// MLPNet_18459769438852
// MI455X (gfx1250) — matrix operand order repaired and checked by reading, not compiled, not run on hardware
//
#include <hip/hip_runtime.h>
#include <hip/hip_bf16.h>
#include <math.h>
#include <stdint.h>

typedef __attribute__((ext_vector_type(16))) _Float16 v16h;
typedef __attribute__((ext_vector_type(8)))  _Float16 v8h;
typedef __attribute__((ext_vector_type(8)))  float    v8f;
typedef __attribute__((ext_vector_type(4)))  unsigned int u32x4;
typedef __attribute__((ext_vector_type(8)))  int      i32x8;
typedef __attribute__((ext_vector_type(4)))  int      i32x4;

// explicit LDS-address-space pointer: guarantees ds_load_* (never flat), 32-bit address
typedef __attribute__((address_space(3))) const v8h* lds_v8p;

#define T_STEPS 100
#define B_SZ    512
#define F_IN    784
#define K1_PAD  800
#define H1_REAL 500
#define H1_PAD  512
#define H2_REAL 100
#define H2_PAD  128
#define O_SZ    10

static __device__ const float KDW   = (float)((1e-9 / (2.5e-8 + 3.5e-8)) * 2e-5); // vel_scale*I_CUR
static __device__ const float X_THf = 1e-7f;
static __device__ const float EPSf  = 1e-5f;

static __device__ __forceinline__ v16h cat8(v8h lo, v8h hi) {
  return __builtin_shufflevector(lo, hi, 0,1,2,3,4,5,6,7,8,9,10,11,12,13,14,15);
}
// load 16B from LDS by byte address -> ds_load_b128
static __device__ __forceinline__ v8h lds_load8(unsigned byte_addr) {
  return *(lds_v8p)(uintptr_t)byte_addr;
}

#if defined(__has_builtin)
#if __has_builtin(__builtin_amdgcn_tensor_load_to_lds) && __has_builtin(__builtin_amdgcn_s_wait_tensorcnt)
#define HAVE_TDM 1
#endif
#endif

#ifdef HAVE_TDM
// Issue a 2D TDM load: tile_d1 rows x tile_d0 f16 elements, row stride stride0 (elems),
// global tile start -> LDS byte address. D# layout per CDNA5 ISA §8.3/§8.4.
static __device__ __forceinline__ void tdm_load_2d_f16(
    const void* gtile, unsigned lds_addr,
    unsigned tile_d0, unsigned tile_d1, unsigned stride0_elems)
{
  unsigned long long ga = (unsigned long long)(uintptr_t)gtile;
  u32x4 g0 = {
    1u,                                                    // count=1 (valid user D#)
    lds_addr,                                              // LDS byte address
    (unsigned)ga,                                          // global_addr[31:0]
    (unsigned)((ga >> 32) & 0x01FFFFFFu) | 0x80000000u     // global_addr[56:32] | type=2
  };
  i32x8 g1 = {
    (int)(1u << 16),                                       // wg_mask=0, data_size=1 (2B)
    (int)((tile_d0 & 0xFFFFu) << 16),                      // tensor_dim0[15:0] (=tile_d0)
    (int)((tile_d0 >> 16) | ((tile_d1 & 0xFFFFu) << 16)),  // tdim0[31:16] | tdim1[15:0]
    (int)((tile_d1 >> 16) | (tile_d0 << 16)),              // tdim1[31:16] | tile_dim0
    (int)(tile_d1 & 0xFFFFu),                              // tile_dim1 | tile_dim2=0
    (int)stride0_elems,                                    // tensor_dim0_stride[31:0]
    0,                                                     // stride0[47:32] | stride1[15:0]
    0
  };
  i32x4 z4 = {0, 0, 0, 0};
#if __clang_major__ >= 23
  i32x8 z8 = {0, 0, 0, 0, 0, 0, 0, 0};
  __builtin_amdgcn_tensor_load_to_lds(g0, g1, z4, z4, z8, 0);
#else
  __builtin_amdgcn_tensor_load_to_lds(g0, g1, z4, z4, 0);
#endif
}
#endif

// ---------------- utility kernels ----------------
__global__ void k_zero(float* __restrict__ p, int n) {
  int i = blockIdx.x * 256 + threadIdx.x;
  if (i < n) p[i] = 0.f;
}

// fc1_w [500x784] f32 -> w1h [512x800] f16, zero-padded
__global__ void k_wconv(const float* __restrict__ fc1_w, _Float16* __restrict__ w1h) {
  int i = blockIdx.x * 256 + threadIdx.x;
  if (i >= H1_PAD * K1_PAD) return;
  int n = i / K1_PAD, k = i % K1_PAD;
  float v = (n < H1_REAL && k < F_IN) ? fc1_w[n * F_IN + k] : 0.f;
  w1h[i] = (_Float16)v;
}

// x_t [512x784] f32 -> xh [512x800] f16, zero-padded (keeps cvt out of the GEMM loop)
__global__ void k_xconv(const float* __restrict__ xt, _Float16* __restrict__ xh) {
  int i = blockIdx.x * 256 + threadIdx.x;
  if (i >= B_SZ * K1_PAD) return;
  int r = i / K1_PAD, k = i % K1_PAD;
  float v = (k < F_IN) ? xt[(size_t)r * F_IN + k] : 0.f;
  xh[i] = (_Float16)v;
}

// ---------------- GEMM1: z1 = x_t @ W1^T + b1, fused dwmtj spike ----------------
// grid (8,4), block 256 (8 waves, 2M x 4N), WG tile 64x128, K = 800
// A/B tiles staged global->LDS by the Tensor Data Mover, double-buffered.
__global__ __launch_bounds__(256)
void k_fc1(const _Float16* __restrict__ xh,      // [512 x 800] f16
           const _Float16* __restrict__ w1h,     // [512 x 800] f16
           const float* __restrict__ fc1_b,      // [500]
           float* __restrict__ x0,               // [512 x 512] state
           _Float16* __restrict__ s1h,           // [512 x 512] spikes (f16)
           float* __restrict__ cnt1)             // [512] spike counts
{
  const int tid  = threadIdx.x;
  const int lane = tid & 31, wave = tid >> 5;
  const int wm = wave >> 2, wn = wave & 3;
  const int l16 = lane & 15, half = lane >> 4;
  const int Mb = blockIdx.x * 64, Nb = blockIdx.y * 128;
  const int NK = K1_PAD / 32;   // 25

  v8f acc[2][2] = {};

#ifdef HAVE_TDM
  __shared__ _Float16 As[2][64 * 32];
  __shared__ _Float16 Bs[2][128 * 32];
  const unsigned ldsA = (unsigned)(uintptr_t)&As[0][0];  // low 32 bits = LDS offset
  const unsigned ldsB = (unsigned)(uintptr_t)&Bs[0][0];

  auto issue = [&](int kk, int buf) {
    const int k0 = kk * 32;
    tdm_load_2d_f16(xh  + (size_t)Mb * K1_PAD + k0, ldsA + buf * (64  * 32 * 2), 32, 64,  K1_PAD);
    tdm_load_2d_f16(w1h + (size_t)Nb * K1_PAD + k0, ldsB + buf * (128 * 32 * 2), 32, 128, K1_PAD);
  };
  if (wave == 0) { issue(0, 0); issue(1, 1); }

  // per-lane LDS byte addresses within a tile
  const unsigned aOff = (unsigned)(((wm * 32 + l16) * 32 + half * 8) * 2);   // +mi*16*64
  const unsigned bOff = (unsigned)(((wn * 32 + l16) * 32 + half * 8) * 2);   // +ni*16*64

  for (int kk = 0; kk < NK; ++kk) {
    const int buf = kk & 1;
    if (wave == 0) {
      if (kk < NK - 1) __builtin_amdgcn_s_wait_tensorcnt(2);  // pair kk complete
      else             __builtin_amdgcn_s_wait_tensorcnt(0);
    }
    __syncthreads();

    v16h a[2], b[2];
    #pragma unroll
    for (int mi = 0; mi < 2; ++mi) {
      unsigned p = ldsA + buf * (64 * 32 * 2) + aOff + mi * (16 * 64);
      a[mi] = cat8(lds_load8(p), lds_load8(p + 32));
    }
    #pragma unroll
    for (int ni = 0; ni < 2; ++ni) {
      unsigned p = ldsB + buf * (128 * 32 * 2) + bOff + ni * (16 * 64);
      b[ni] = cat8(lds_load8(p), lds_load8(p + 32));
    }
    #pragma unroll
    for (int mi = 0; mi < 2; ++mi)
      #pragma unroll
      for (int ni = 0; ni < 2; ++ni)
        acc[mi][ni] = __builtin_amdgcn_wmma_f32_16x16x32_f16(
            false, a[mi], false, b[ni], (short)0, acc[mi][ni], false, false);

    __syncthreads();                       // all frag reads of buf done
    if (wave == 0 && kk + 2 < NK) issue(kk + 2, buf);
  }
#else
  for (int k0 = 0; k0 < K1_PAD; k0 += 32) {
    v16h a[2], b[2];
    #pragma unroll
    for (int mi = 0; mi < 2; ++mi) {
      const _Float16* p = xh + (size_t)(Mb + wm * 32 + mi * 16 + l16) * K1_PAD + k0 + half * 8;
      a[mi] = cat8(*(const v8h*)p, *(const v8h*)(p + 16));
    }
    #pragma unroll
    for (int ni = 0; ni < 2; ++ni) {
      const _Float16* p = w1h + (size_t)(Nb + wn * 32 + ni * 16 + l16) * K1_PAD + k0 + half * 8;
      b[ni] = cat8(*(const v8h*)p, *(const v8h*)(p + 16));
    }
    #pragma unroll
    for (int mi = 0; mi < 2; ++mi)
      #pragma unroll
      for (int ni = 0; ni < 2; ++ni)
        acc[mi][ni] = __builtin_amdgcn_wmma_f32_16x16x32_f16(
            false, a[mi], false, b[ni], (short)0, acc[mi][ni], false, false);
  }
#endif

  // epilogue: bias + domain-wall integrate + spike + reset + counts
  #pragma unroll
  for (int mi = 0; mi < 2; ++mi)
    #pragma unroll
    for (int ni = 0; ni < 2; ++ni) {
      int gn = Nb + wn * 32 + ni * 16 + l16;
      bool valid = gn < H1_REAL;
      float bias = valid ? fc1_b[gn] : 0.f;
      float csum = 0.f;
      #pragma unroll
      for (int r = 0; r < 8; ++r) {
        int gm = Mb + wm * 32 + mi * 16 + r + half * 8;
        float s = 0.f;
        if (valid) {
          float z  = acc[mi][ni][r] + bias;
          float xp = x0[gm * H1_PAD + gn] + KDW * z;
          s = (xp > X_THf) ? 1.f : 0.f;
          x0[gm * H1_PAD + gn] = (s > 0.f) ? 0.f : xp;
        }
        s1h[gm * H1_PAD + gn] = (_Float16)s;
        csum += s;
      }
      if (valid && csum > 0.f) atomicAdd(cnt1 + gn, csum);
    }
}

// ---------------- stats1: BN1 affine fold + fc2 weight premul + d2 ----------------
__global__ __launch_bounds__(512)
void k_stats1(const float* __restrict__ bn1_g, const float* __restrict__ bn1_b,
              const float* __restrict__ fc2_w, const float* __restrict__ fc2_b,
              float* __restrict__ a1, float* __restrict__ c1, float* __restrict__ d2,
              _Float16* __restrict__ w2p, float* __restrict__ cnt1)
{
  int tid = threadIdx.x;
  {
    float p = cnt1[tid] * (1.f / 512.f);
    float A = 0.f, C = 0.f;
    if (tid < H1_REAL) {
      float var = p - p * p;                 // binary spikes: exact biased variance
      float rs  = 1.f / sqrtf(var + EPSf);
      A = bn1_g[tid] * rs;
      C = bn1_b[tid] - A * p;
    }
    a1[tid] = A; c1[tid] = C;
    cnt1[tid] = 0.f;                         // reset for next step
  }
  __syncthreads();
  for (int i = tid; i < H2_PAD * H1_PAD; i += 512) {
    int j = i >> 9, k = i & 511;
    float v = 0.f;
    if (j < H2_REAL && k < H1_REAL) v = fc2_w[j * H1_REAL + k] * a1[k];
    w2p[i] = (_Float16)v;
  }
  if (tid < H2_PAD) {
    float s = 0.f;
    if (tid < H2_REAL) {
      s = fc2_b[tid];
      for (int k = 0; k < H1_REAL; ++k) s += c1[k] * fc2_w[tid * H1_REAL + k];
    }
    d2[tid] = s;
  }
}

// ---------------- GEMM2: z2 = s1 @ (a1*W2)^T + d2, fused dwmtj spike ----------------
__global__ __launch_bounds__(256)
void k_fc2(const _Float16* __restrict__ s1h,   // [512 x 512]
           const _Float16* __restrict__ w2p,   // [128 x 512]
           const float* __restrict__ d2,       // [128]
           float* __restrict__ x1,             // [512 x 128] state
           _Float16* __restrict__ s2h,         // [512 x 128]
           float* __restrict__ cnt2)           // [128]
{
  const int tid  = threadIdx.x;
  const int lane = tid & 31, wave = tid >> 5;
  const int wm = wave >> 2, wn = wave & 3;
  const int l16 = lane & 15, half = lane >> 4;
  const int Mb = blockIdx.x * 64;

  v8f acc[2][2] = {};

  for (int k0 = 0; k0 < H1_PAD; k0 += 32) {
    v16h a[2], b[2];
    #pragma unroll
    for (int mi = 0; mi < 2; ++mi) {
      const _Float16* p = s1h + (size_t)(Mb + wm * 32 + mi * 16 + l16) * H1_PAD + k0 + half * 8;
      a[mi] = cat8(*(const v8h*)p, *(const v8h*)(p + 16));
    }
    #pragma unroll
    for (int ni = 0; ni < 2; ++ni) {
      const _Float16* p = w2p + (size_t)(wn * 32 + ni * 16 + l16) * H1_PAD + k0 + half * 8;
      b[ni] = cat8(*(const v8h*)p, *(const v8h*)(p + 16));
    }
    #pragma unroll
    for (int mi = 0; mi < 2; ++mi)
      #pragma unroll
      for (int ni = 0; ni < 2; ++ni)
        acc[mi][ni] = __builtin_amdgcn_wmma_f32_16x16x32_f16(
            false, a[mi], false, b[ni], (short)0, acc[mi][ni], false, false);
  }

  #pragma unroll
  for (int mi = 0; mi < 2; ++mi)
    #pragma unroll
    for (int ni = 0; ni < 2; ++ni) {
      int gn = wn * 32 + ni * 16 + l16;
      bool valid = gn < H2_REAL;
      float dj = valid ? d2[gn] : 0.f;
      float csum = 0.f;
      #pragma unroll
      for (int r = 0; r < 8; ++r) {
        int gm = Mb + wm * 32 + mi * 16 + r + half * 8;
        float s = 0.f;
        if (valid) {
          float z  = acc[mi][ni][r] + dj;
          float xp = x1[gm * H2_PAD + gn] + KDW * z;
          s = (xp > X_THf) ? 1.f : 0.f;
          x1[gm * H2_PAD + gn] = (s > 0.f) ? 0.f : xp;
        }
        s2h[gm * H2_PAD + gn] = (_Float16)s;   // relu(s)=s, binary
        csum += s;
      }
      if (valid && csum > 0.f) atomicAdd(cnt2 + gn, csum);
    }
}

// ---------------- stats2: BN2 fold + out_w premul + e ----------------
__global__ __launch_bounds__(128)
void k_stats2(const float* __restrict__ bn2_g, const float* __restrict__ bn2_b,
              const float* __restrict__ out_w,
              float* __restrict__ a2, float* __restrict__ c2, float* __restrict__ e,
              float* __restrict__ w3p, float* __restrict__ cnt2)
{
  int tid = threadIdx.x;
  {
    float p = cnt2[tid] * (1.f / 512.f);
    float A = 0.f, C = 0.f;
    if (tid < H2_REAL) {
      float var = p - p * p;
      float rs  = 1.f / sqrtf(var + EPSf);
      A = bn2_g[tid] * rs;
      C = bn2_b[tid] - A * p;
    }
    a2[tid] = A; c2[tid] = C;
    cnt2[tid] = 0.f;
  }
  __syncthreads();
  for (int i = tid; i < O_SZ * H2_REAL; i += 128) {
    int j = i % H2_REAL;
    w3p[i] = out_w[i] * a2[j];
  }
  if (tid < O_SZ) {
    float s = 0.f;
    for (int j = 0; j < H2_REAL; ++j) s += c2[j] * out_w[tid * H2_REAL + j];
    e[tid] = s;
  }
}

// ---------------- LI output cell ----------------
__global__ __launch_bounds__(256)
void k_out(const _Float16* __restrict__ s2h, const float* __restrict__ w3p,
           const float* __restrict__ e, float* __restrict__ vst, float* __restrict__ ist,
           float* __restrict__ out_t)
{
  int id = blockIdx.x * 256 + threadIdx.x;   // < 5120
  if (id >= B_SZ * O_SZ) return;
  int b = id / O_SZ, o = id % O_SZ;
  float dot = e[o];
  const _Float16* sp = s2h + b * H2_PAD;
  for (int j = 0; j < H2_REAL; ++j) dot += (float)sp[j] * w3p[o * H2_REAL + j];
  float ij = ist[id] + dot;                  // i + z@out_w^T (+c-term)
  float v  = vst[id];
  float vn = v + 0.1f * (ij - v);            // DT_LI*TAU_MEM_INV = 0.1
  ist[id]  = 0.8f * ij;                      // 1 - DT_LI*TAU_SYN_INV
  vst[id]  = vn;
  out_t[id] = vn;
}

extern "C" void kernel_launch(void* const* d_in, const int* in_sizes, int n_in,
                              void* d_out, int out_size, void* d_ws, size_t ws_size,
                              hipStream_t stream) {
  (void)in_sizes; (void)n_in; (void)out_size; (void)ws_size;
  const float* x     = (const float*)d_in[0];
  const float* fc1_w = (const float*)d_in[1];
  const float* fc1_b = (const float*)d_in[2];
  const float* fc2_w = (const float*)d_in[3];
  const float* fc2_b = (const float*)d_in[4];
  const float* bn1_g = (const float*)d_in[5];
  const float* bn1_b = (const float*)d_in[6];
  const float* bn2_g = (const float*)d_in[7];
  const float* bn2_b = (const float*)d_in[8];
  const float* out_w = (const float*)d_in[9];
  float* out = (float*)d_out;

  char* ws = (char*)d_ws;
  size_t off = 0;
  auto carve = [&](size_t bytes) -> char* {
    char* p = ws + off;
    off += (bytes + 255) & ~(size_t)255;
    return p;
  };
  _Float16* w1h = (_Float16*)carve((size_t)H1_PAD * K1_PAD * 2);
  _Float16* xh  = (_Float16*)carve((size_t)B_SZ * K1_PAD * 2);
  _Float16* s1h = (_Float16*)carve((size_t)B_SZ * H1_PAD * 2);
  _Float16* w2p = (_Float16*)carve((size_t)H2_PAD * H1_PAD * 2);
  _Float16* s2h = (_Float16*)carve((size_t)B_SZ * H2_PAD * 2);

  // contiguous zero-init state region
  const int ZN = B_SZ * H1_PAD + B_SZ * H2_PAD + B_SZ * O_SZ * 2 + H1_PAD + H2_PAD;
  float* zbase = (float*)carve((size_t)ZN * 4);
  float* x0   = zbase;
  float* x1   = x0 + B_SZ * H1_PAD;
  float* vst  = x1 + B_SZ * H2_PAD;
  float* ist  = vst + B_SZ * O_SZ;
  float* cnt1 = ist + B_SZ * O_SZ;
  float* cnt2 = cnt1 + H1_PAD;

  float* a1  = (float*)carve(H1_PAD * 4);
  float* c1  = (float*)carve(H1_PAD * 4);
  float* d2  = (float*)carve(H2_PAD * 4);
  float* a2  = (float*)carve(H2_PAD * 4);
  float* c2  = (float*)carve(H2_PAD * 4);
  float* w3p = (float*)carve(O_SZ * H2_REAL * 4);
  float* e   = (float*)carve(64);

  k_zero<<<(ZN + 255) / 256, 256, 0, stream>>>(zbase, ZN);
  k_wconv<<<(H1_PAD * K1_PAD + 255) / 256, 256, 0, stream>>>(fc1_w, w1h);

  for (int t = 0; t < T_STEPS; ++t) {
    const float* xt = x + (size_t)t * B_SZ * F_IN;
    k_xconv<<<(B_SZ * K1_PAD + 255) / 256, 256, 0, stream>>>(xt, xh);
    k_fc1<<<dim3(8, 4), 256, 0, stream>>>(xh, w1h, fc1_b, x0, s1h, cnt1);
    k_stats1<<<1, 512, 0, stream>>>(bn1_g, bn1_b, fc2_w, fc2_b, a1, c1, d2, w2p, cnt1);
    k_fc2<<<dim3(8, 1), 256, 0, stream>>>(s1h, w2p, d2, x1, s2h, cnt2);
    k_stats2<<<1, 128, 0, stream>>>(bn2_g, bn2_b, out_w, a2, c2, e, w3p, cnt2);
    k_out<<<(B_SZ * O_SZ + 255) / 256, 256, 0, stream>>>(s2h, w3p, e, vst, ist,
                                                         out + (size_t)t * B_SZ * O_SZ);
  }
}
